// AttentionBlock_9783935500353
// MI455X (gfx1250) — matrix operand order repaired and checked by reading, not compiled, not run on hardware
//
#include <hip/hip_runtime.h>

typedef __attribute__((ext_vector_type(16))) __bf16 v16bf;
typedef __attribute__((ext_vector_type(8)))  __bf16 v8bf;
typedef __attribute__((ext_vector_type(8)))  float  v8f;

#define B_  4
#define C_  256
#define IC_ 128
#define N_  4096

__device__ __forceinline__ v8f wmma_bf16(v16bf a, v16bf b, v8f c) {
    return __builtin_amdgcn_wmma_f32_16x16x32_bf16(false, a, false, b, (short)0, c, false, false);
}

// ---- async global->LDS staging (gfx1250), guarded with fallback ------------
#if __has_builtin(__builtin_amdgcn_global_load_async_to_lds_b128)
#define USE_ASYNC 1
typedef int v4i_ __attribute__((vector_size(16)));
typedef __attribute__((address_space(1))) v4i_* g4p_t;   // global int4*
typedef __attribute__((address_space(3))) v4i_* l4p_t;   // LDS int4*
#define TO_GLB(p) ((g4p_t)(unsigned long long)(p))
#define TO_LDS(p) ((l4p_t)(unsigned int)(unsigned long long)(p))
#if __has_builtin(__builtin_amdgcn_s_wait_asynccnt)
#define WAIT_ASYNC_8() __builtin_amdgcn_s_wait_asynccnt(8)
#define WAIT_ASYNC_0() __builtin_amdgcn_s_wait_asynccnt(0)
#else
#define WAIT_ASYNC_8() asm volatile("s_wait_asynccnt 8" ::: "memory")
#define WAIT_ASYNC_0() asm volatile("s_wait_asynccnt 0" ::: "memory")
#endif
#else
#define USE_ASYNC 0
#define WAIT_ASYNC_8()
#define WAIT_ASYNC_0()
#endif

// ---------------------------------------------------------------------------
// Kernel 1: three input 1x1 convs.
//   theta,g : out[b][ic][n]  (bf16)
//   phi     : out[b][n][ic]  (bf16, TRANSPOSED via LDS bounce for fast attn frags)
// grid = (B_*256, 3), block = 128 (4 waves); wave w computes ic-tiles {2w,2w+1}.
// ---------------------------------------------------------------------------
__global__ void __launch_bounds__(128) conv_in_kernel(
    const float* __restrict__ x, const float* __restrict__ y,
    const float* __restrict__ dxw, const float* __restrict__ dxb,
    const float* __restrict__ dyw, const float* __restrict__ dyb,
    const float* __restrict__ gw,  const float* __restrict__ gb,
    __bf16* __restrict__ theta, __bf16* __restrict__ phiT, __bf16* __restrict__ g)
{
    __shared__ float tr[IC_ * 16];

    const int tid  = threadIdx.x;
    const int lane = tid & 31;
    const int wave = tid >> 5;
    const int col  = lane & 15;
    const int half = lane >> 4;

    const int b  = blockIdx.x >> 8;
    const int n0 = (blockIdx.x & 255) * 16;

    const float* in; const float* wmat; const float* bias; __bf16* out;
    if (blockIdx.y == 0)      { in = x; wmat = dxw; bias = dxb; out = theta; }
    else if (blockIdx.y == 1) { in = y; wmat = dyw; bias = dyb; out = phiT; }
    else                      { in = y; wmat = gw;  bias = gb;  out = g; }

    v8f acc0 = {}; v8f acc1 = {};
    for (int kc = 0; kc < 8; ++kc) {
        // B fragment from input: lane col = n, e -> c = kc*32 + 16*half + e (stride N in mem)
        v16bf bf;
        {
            const float* p = in + ((size_t)b * C_ + kc * 32 + half * 16) * N_ + n0 + col;
#pragma unroll
            for (int e = 0; e < 16; ++e) bf[e] = (__bf16)p[(size_t)e * N_];
        }
        // A fragments from weight: lane row = ic, e -> c = kc*32 + 16*half + e
#pragma unroll
        for (int t = 0; t < 2; ++t) {
            const int ic = (wave * 2 + t) * 16 + col;
            const float* p = wmat + (size_t)ic * C_ + kc * 32 + half * 16;
            v16bf af;
#pragma unroll
            for (int e = 0; e < 16; ++e) af[e] = (__bf16)p[e];
            if (t == 0) acc0 = wmma_bf16(af, bf, acc0);
            else        acc1 = wmma_bf16(af, bf, acc1);
        }
    }

    if (blockIdx.y != 1) {
        // plain [ic][n] store
#pragma unroll
        for (int t = 0; t < 2; ++t) {
            const v8f a = (t == 0) ? acc0 : acc1;
#pragma unroll
            for (int r = 0; r < 8; ++r) {
                const int ic = (wave * 2 + t) * 16 + r + half * 8;
                out[((size_t)b * IC_ + ic) * N_ + n0 + col] = (__bf16)(a[r] + bias[ic]);
            }
        }
    } else {
        // transpose via LDS: tr[ic][n_local] then write [n][ic] rows coalesced
#pragma unroll
        for (int t = 0; t < 2; ++t) {
            const v8f a = (t == 0) ? acc0 : acc1;
#pragma unroll
            for (int r = 0; r < 8; ++r) {
                const int ic = (wave * 2 + t) * 16 + r + half * 8;
                tr[ic * 16 + col] = a[r] + bias[ic];
            }
        }
        __syncthreads();
        const int row = tid >> 3;   // 0..15 (n within tile)
        const int seg = tid & 7;    // 16 ic each
        v16bf v;
#pragma unroll
        for (int e = 0; e < 16; ++e) v[e] = (__bf16)tr[(seg * 16 + e) * 16 + row];
        *(v16bf*)(out + ((size_t)b * N_ + n0 + row) * IC_ + seg * 16) = v;
    }
}

// ---------------------------------------------------------------------------
// Kernel 2: fused attention (flash style).  y2[b][n][ic] (bf16)
// grid = B_ * (N_/64), block = 128; wave w owns rows [rt*64 + 16w, +16).
// Double-buffered phi/g chunk staging (async global->LDS when available).
// ---------------------------------------------------------------------------
__global__ void __launch_bounds__(128) attn_kernel(
    const __bf16* __restrict__ theta, const __bf16* __restrict__ phiT,
    const __bf16* __restrict__ g, __bf16* __restrict__ y2)
{
    __shared__ __bf16 phiS[2][32 * IC_];   // [n_local][ic]
    __shared__ __bf16 gS[2][IC_ * 32];     // [ic][m_local]
    __shared__ __bf16 Pbuf[4][16 * 32];

    const int tid  = threadIdx.x;
    const int lane = tid & 31;
    const int wave = tid >> 5;
    const int col  = lane & 15;
    const int half = lane >> 4;

    const int b    = blockIdx.x >> 6;
    const int row0 = (blockIdx.x & 63) * 64 + wave * 16;

    const __bf16* thetaB = theta + (size_t)b * IC_ * N_;
    const __bf16* phiTB  = phiT  + (size_t)b * N_ * IC_;
    const __bf16* gB     = g     + (size_t)b * IC_ * N_;

    // theta A-fragments (loaded once): lane row = row0+col, e -> ic
    v16bf afr[4];
#pragma unroll
    for (int kc = 0; kc < 4; ++kc) {
        const int icb = kc * 32 + half * 16;
#pragma unroll
        for (int e = 0; e < 16; ++e)
            afr[kc][e] = thetaB[(size_t)(icb + e) * N_ + row0 + col];
    }

    v8f O[8];
    float m[8], l[8];
#pragma unroll
    for (int t = 0; t < 8; ++t) O[t] = (v8f){0.f,0.f,0.f,0.f,0.f,0.f,0.f,0.f};
#pragma unroll
    for (int r = 0; r < 8; ++r) { m[r] = -1e30f; l[r] = 0.f; }

    const int prow = tid >> 2, pseg = tid & 3;   // phi staging: 32 rows x 4 segs of 64B
    auto stage = [&](int nc, int bi) {
        const __bf16* psrc = phiTB + (size_t)(nc + prow) * IC_ + pseg * 32;
        const __bf16* gsrc = gB + (size_t)tid * N_ + nc;
        __bf16* pdst = &phiS[bi][prow * IC_ + pseg * 32];
        __bf16* gdst = &gS[bi][tid * 32];
#if USE_ASYNC
#pragma unroll
        for (int j = 0; j < 4; ++j) {
            __builtin_amdgcn_global_load_async_to_lds_b128(TO_GLB(psrc + j * 8), TO_LDS(pdst + j * 8), 0, 0);
            __builtin_amdgcn_global_load_async_to_lds_b128(TO_GLB(gsrc + j * 8), TO_LDS(gdst + j * 8), 0, 0);
        }
#else
#pragma unroll
        for (int j = 0; j < 4; ++j) {
            ((uint4*)pdst)[j] = ((const uint4*)psrc)[j];
            ((uint4*)gdst)[j] = ((const uint4*)gsrc)[j];
        }
        if (nc + 32 < N_) {
            __builtin_prefetch(psrc + IC_ * 32, 0, 1);
            __builtin_prefetch(gsrc + 32, 0, 1);
        }
#endif
    };

    stage(0, 0);
    for (int it = 0; it < N_ / 32; ++it) {
        const int bi = it & 1;
        if (it + 1 < N_ / 32) { stage((it + 1) * 32, bi ^ 1); WAIT_ASYNC_8(); }
        else                  { WAIT_ASYNC_0(); }
        __syncthreads();

        const __bf16* pS = &phiS[bi][0];
        const __bf16* gSb = &gS[bi][0];

        // S = theta_tile (16x128) @ phi_chunk (128x32): two 16x16 f32 tiles
        v8f S0 = {}; v8f S1 = {};
#pragma unroll
        for (int kc = 0; kc < 4; ++kc) {
            const v16bf bf0 = *(const v16bf*)(pS + (size_t)col * IC_ + kc * 32 + half * 16);
            const v16bf bf1 = *(const v16bf*)(pS + (size_t)(16 + col) * IC_ + kc * 32 + half * 16);
            S0 = wmma_bf16(afr[kc], bf0, S0);
            S1 = wmma_bf16(afr[kc], bf1, S1);
        }

        // online softmax (rows live at r + 8*half; 16-lane reductions per half)
        float mnew[8], alpha[8];
#pragma unroll
        for (int r = 0; r < 8; ++r) {
            float t = fmaxf(S0[r], S1[r]);
            t = fmaxf(t, __shfl_xor(t, 1, 32));
            t = fmaxf(t, __shfl_xor(t, 2, 32));
            t = fmaxf(t, __shfl_xor(t, 4, 32));
            t = fmaxf(t, __shfl_xor(t, 8, 32));
            mnew[r]  = fmaxf(m[r], t);
            alpha[r] = __expf(m[r] - mnew[r]);
            m[r]     = mnew[r];
        }
#pragma unroll
        for (int r = 0; r < 8; ++r) {
            const float p0 = __expf(S0[r] - mnew[r]);
            const float p1 = __expf(S1[r] - mnew[r]);
            S0[r] = p0; S1[r] = p1;
            float rs = p0 + p1;
            rs += __shfl_xor(rs, 1, 32);
            rs += __shfl_xor(rs, 2, 32);
            rs += __shfl_xor(rs, 4, 32);
            rs += __shfl_xor(rs, 8, 32);
            l[r] = l[r] * alpha[r] + rs;
        }

        // transpose P (C/D layout) -> A layout via per-wave LDS bounce
        __bf16* pb = &Pbuf[wave][0];
#pragma unroll
        for (int r = 0; r < 8; ++r) {
            pb[(r + half * 8) * 32 + col]      = (__bf16)S0[r];
            pb[(r + half * 8) * 32 + 16 + col] = (__bf16)S1[r];
        }
        asm volatile("s_wait_dscnt 0" ::: "memory");
        v16bf pf;
        {
            const v8bf lo = *(const v8bf*)(pb + col * 32 + half * 16);
            const v8bf hi = *(const v8bf*)(pb + col * 32 + half * 16 + 8);
            pf = __builtin_shufflevector(lo, hi, 0,1,2,3,4,5,6,7,8,9,10,11,12,13,14,15);
        }

        // O = O*alpha + P (16x32) @ g_chunk (32x128)
#pragma unroll
        for (int t = 0; t < 8; ++t) {
#pragma unroll
            for (int r = 0; r < 8; ++r) O[t][r] *= alpha[r];
        }
#pragma unroll
        for (int t = 0; t < 8; ++t) {
            const v16bf gf = *(const v16bf*)(gSb + (t * 16 + col) * 32 + half * 16);
            O[t] = wmma_bf16(pf, gf, O[t]);
        }
        __syncthreads();
    }

    // epilogue: normalize rows, store y2[b][n][ic] bf16
#pragma unroll
    for (int r = 0; r < 8; ++r) {
        const float inv = 1.0f / l[r];
        const int n = row0 + r + half * 8;
#pragma unroll
        for (int t = 0; t < 8; ++t)
            y2[((size_t)b * N_ + n) * IC_ + t * 16 + col] = (__bf16)(O[t][r] * inv);
    }
}

// ---------------------------------------------------------------------------
// Kernel 3: output conv.  wy[b][c][n] = sum_ic W[c][ic]*y2[b][n][ic] + wb[c]  (f32)
// ---------------------------------------------------------------------------
__global__ void __launch_bounds__(128) conv_out_kernel(
    const __bf16* __restrict__ y2, const float* __restrict__ ww,
    const float* __restrict__ wb, float* __restrict__ wy)
{
    const int lane = threadIdx.x & 31;
    const int wave = threadIdx.x >> 5;
    const int col  = lane & 15;
    const int half = lane >> 4;
    const int b    = blockIdx.x >> 8;
    const int n0   = (blockIdx.x & 255) * 16;

    const __bf16* y2B = y2 + (size_t)b * N_ * IC_;
    v8f acc[4] = {};
    for (int kc = 0; kc < 4; ++kc) {
        const v16bf bf = *(const v16bf*)(y2B + (size_t)(n0 + col) * IC_ + kc * 32 + half * 16);
#pragma unroll
        for (int t = 0; t < 4; ++t) {
            const int c = (wave * 4 + t) * 16 + col;
            const float* p = ww + (size_t)c * IC_ + kc * 32 + half * 16;
            v16bf af;
#pragma unroll
            for (int e = 0; e < 16; ++e) af[e] = (__bf16)p[e];
            acc[t] = wmma_bf16(af, bf, acc[t]);
        }
    }
#pragma unroll
    for (int t = 0; t < 4; ++t) {
#pragma unroll
        for (int r = 0; r < 8; ++r) {
            const int c = (wave * 4 + t) * 16 + r + half * 8;
            wy[((size_t)b * C_ + c) * N_ + n0 + col] = acc[t][r] + wb[c];
        }
    }
}

// ---------------------------------------------------------------------------
// Kernel 4: per-channel batch stats over (B,N).  stats[c] = {mean, var}
// ---------------------------------------------------------------------------
__global__ void stats_kernel(const float* __restrict__ wy, float* __restrict__ stats)
{
    __shared__ float s1[256], s2[256];
    const int c = blockIdx.x;
    float a = 0.f, q = 0.f;
    for (int b = 0; b < B_; ++b) {
        const float* p = wy + ((size_t)b * C_ + c) * N_;
        for (int n = threadIdx.x; n < N_; n += 256) { const float v = p[n]; a += v; q += v * v; }
    }
    s1[threadIdx.x] = a; s2[threadIdx.x] = q;
    __syncthreads();
    for (int s = 128; s > 0; s >>= 1) {
        if (threadIdx.x < s) { s1[threadIdx.x] += s1[threadIdx.x + s]; s2[threadIdx.x] += s2[threadIdx.x + s]; }
        __syncthreads();
    }
    if (threadIdx.x == 0) {
        const float inv = 1.0f / (float)(B_ * N_);
        const float mean = s1[0] * inv;
        stats[c * 2]     = mean;
        stats[c * 2 + 1] = s2[0] * inv - mean * mean;
    }
}

// ---------------------------------------------------------------------------
// Kernel 5: BN apply + residual.
// ---------------------------------------------------------------------------
__global__ void bn_apply_kernel(
    const float* __restrict__ wy, const float* __restrict__ x,
    const float* __restrict__ gamma, const float* __restrict__ beta,
    const float* __restrict__ stats, float* __restrict__ out)
{
    const size_t i = (size_t)blockIdx.x * 256 + threadIdx.x;
    if (i >= (size_t)B_ * C_ * N_) return;
    const int c = (int)((i / N_) % C_);
    const float mean = stats[2 * c];
    const float var  = stats[2 * c + 1];
    out[i] = (wy[i] - mean) * rsqrtf(var + 1e-5f) * gamma[c] + beta[c] + x[i];
}

// ---------------------------------------------------------------------------
extern "C" void kernel_launch(void* const* d_in, const int* in_sizes, int n_in,
                              void* d_out, int out_size, void* d_ws, size_t ws_size,
                              hipStream_t stream) {
    const float* x     = (const float*)d_in[0];
    const float* y     = (const float*)d_in[1];
    const float* g_w   = (const float*)d_in[2];
    const float* g_b   = (const float*)d_in[3];
    const float* dx_w  = (const float*)d_in[4];
    const float* dx_b  = (const float*)d_in[5];
    const float* dy_w  = (const float*)d_in[6];
    const float* dy_b  = (const float*)d_in[7];
    const float* w_w   = (const float*)d_in[8];
    const float* w_b   = (const float*)d_in[9];
    const float* bn_g  = (const float*)d_in[10];
    const float* bn_b  = (const float*)d_in[11];

    const size_t act = (size_t)B_ * IC_ * N_;
    __bf16* theta = (__bf16*)d_ws;
    __bf16* phiT  = theta + act;
    __bf16* g     = phiT + act;
    __bf16* y2    = g + act;
    float*  wy    = (float*)(y2 + act);
    float*  stats = wy + (size_t)B_ * C_ * N_;

    conv_in_kernel<<<dim3(B_ * 256, 3), 128, 0, stream>>>(
        x, y, dx_w, dx_b, dy_w, dy_b, g_w, g_b, theta, phiT, g);

    attn_kernel<<<B_ * (N_ / 64), 128, 0, stream>>>(theta, phiT, g, y2);

    conv_out_kernel<<<B_ * 256, 128, 0, stream>>>(y2, w_w, w_b, wy);

    stats_kernel<<<C_, 256, 0, stream>>>(wy, stats);

    const size_t total = (size_t)B_ * C_ * N_;
    bn_apply_kernel<<<(unsigned)((total + 255) / 256), 256, 0, stream>>>(
        wy, x, bn_g, bn_b, stats, (float*)d_out);
}
